// Seq2seq_with_attention_73907797229584
// MI455X (gfx1250) — matrix operand order repaired and checked by reading, not compiled, not run on hardware
//
#include <hip/hip_runtime.h>
#include <hip/hip_bf16.h>

// ---------------------------------------------------------------------------
// CDNA5 (gfx1250) seq2seq-with-attention forward, bf16 WMMA (f32 accumulate).
// Model: V=32000, E=H=512, FF=2048, L=2, B=32, S=64, T=48.
// All WMMA operands pre-packed to bf16; inner loops are pure b128-load + wmma.
// ---------------------------------------------------------------------------

typedef __attribute__((ext_vector_type(16))) __bf16 v16bf;
typedef __attribute__((ext_vector_type(8)))  float  v8f;
typedef unsigned short ushort_t;

#define DEVINL static __device__ __forceinline__

DEVINL ushort_t f2bf(float f) {
  union { float f; unsigned u; } c; c.f = f;
  unsigned u = c.u;
  u += 0x7FFFu + ((u >> 16) & 1u);          // round-to-nearest-even
  return (ushort_t)(u >> 16);
}

DEVINL float bf2f(ushort_t s) {
  union { unsigned u; float f; } c; c.u = ((unsigned)s) << 16;
  return c.f;
}

union BF16Frag { v16bf v; uint4 q[2]; ushort_t u[16]; };

DEVINL v8f zero8() {
  v8f z;
#pragma unroll
  for (int j = 0; j < 8; ++j) z[j] = 0.f;
  return z;
}

DEVINL v8f wmma32(v16bf a, v16bf b, v8f c) {
  // D = A(16x32 bf16) x B(32x16 bf16) + C(16x16 f32)
  return __builtin_amdgcn_wmma_f32_16x16x32_bf16(false, a, false, b,
                                                 (short)0, c, false, false);
}

// A fragment: weight tile row nRow of bf16 W[N,K] (row stride lda elements).
// frag elems 0..7  = K(kc + khalf*16 + 0..7)      -> one 16B load
// frag elems 8..15 = K(kc + khalf*16 + 8..15)     -> one 16B load
DEVINL v16bf load_a_bf16(const ushort_t* __restrict__ W, int lda, int nRow,
                         int kc, int khalf) {
  const ushort_t* p = W + (size_t)nRow * lda + kc + khalf * 16;
  BF16Frag r;
  r.q[0] = *(const uint4*)(p);
  r.q[1] = *(const uint4*)(p + 8);
  return r.v;
}

// B fragment: B[k][m] = X[mRow][k]; 16 consecutive K starting kc+khalf*16.
DEVINL v16bf load_b_bf16(const ushort_t* __restrict__ X, int ldb, int mRow,
                         int kc, int khalf) {
  const ushort_t* p = X + (size_t)mRow * ldb + kc + khalf * 16;
  BF16Frag r;
  r.q[0] = *(const uint4*)(p);
  r.q[1] = *(const uint4*)(p + 8);
  return r.v;
}

DEVINL v16bf load_b_lds(const ushort_t* H, int ldb, int mRow,
                        int kc, int khalf) {
  const ushort_t* p = H + mRow * ldb + kc + khalf * 16;
  BF16Frag r;
  r.q[0] = *(const uint4*)(p);
  r.q[1] = *(const uint4*)(p + 8);
  return r.v;
}

DEVINL float sigm(float x) { return 1.f / (1.f + expf(-x)); }

// ---------------------------------------------------------------------------
// f32 -> packed bf16 convert (one-shot, bandwidth bound)
// ---------------------------------------------------------------------------
__global__ void k_cvt(const float* __restrict__ src, ushort_t* __restrict__ dst,
                      int n) {
  int idx = blockIdx.x * blockDim.x + threadIdx.x;
  if (idx < n) dst[idx] = f2bf(src[idx]);
}

// ---------------------------------------------------------------------------
// Generic GEMM: C[M,N] = X[M,K] @ W[N,K]^T (+bias) (+relu)
// Wave computes 32(M) x 16(N): 2 WMMA tiles sharing the weight fragment.
// mode: 0 row-major; 1 enc_out [(b*64+s),N]; 2 classifier out[b,t+1,n].
// outBf16: store packed bf16 instead of f32.
// ---------------------------------------------------------------------------
__global__ void k_gemm(const ushort_t* __restrict__ W, int lda,
                       const ushort_t* __restrict__ X, int ldb,
                       const float* __restrict__ bias,
                       void* __restrict__ Cout,
                       int M, int N, int K, int relu, int mode, int outBf16) {
  int wave = threadIdx.x >> 5;
  int lane = threadIdx.x & 31;
  int task = blockIdx.x * (blockDim.x >> 5) + wave;
  int mPairs = M >> 5, nTiles = N >> 4;
  if (task >= mPairs * nTiles) return;      // wave-uniform exit
  int pm = task % mPairs;
  int nt = task / mPairs;
  int khalf = lane >> 4, lr = lane & 15;
  int nRow = nt * 16 + lr;
  int mRow0 = pm * 32 + lr;
  v8f acc0 = zero8(), acc1 = zero8();
  for (int kc = 0; kc < K; kc += 32) {
    v16bf a  = load_a_bf16(W, lda, nRow, kc, khalf);
    v16bf b0 = load_b_bf16(X, ldb, mRow0, kc, khalf);
    v16bf b1 = load_b_bf16(X, ldb, mRow0 + 16, kc, khalf);
    acc0 = wmma32(a, b0, acc0);
    acc1 = wmma32(a, b1, acc1);
  }
  int nBase = nt * 16 + khalf * 8;
#pragma unroll
  for (int half = 0; half < 2; ++half) {
    int mRow = mRow0 + half * 16;
    v8f acc = half ? acc1 : acc0;
#pragma unroll
    for (int j = 0; j < 8; ++j) {
      int n = nBase + j;
      float v = acc[j] + (bias ? bias[n] : 0.f);
      if (relu) v = v > 0.f ? v : 0.f;
      size_t oidx;
      if (mode == 0) {
        oidx = (size_t)mRow * N + n;
      } else if (mode == 1) {               // m = s*32+b -> [(b*64+s), n]
        int s = mRow >> 5, b = mRow & 31;
        oidx = (size_t)((b << 6) + s) * N + n;
      } else {                              // m = t*32+b -> out[b, t+1, n]
        int t = mRow >> 5, b = mRow & 31;
        oidx = (size_t)b * 1536000u + (size_t)(t + 1) * 32000u + (size_t)n;
      }
      if (outBf16) ((ushort_t*)Cout)[oidx] = f2bf(v);
      else         ((float*)Cout)[oidx] = v;
    }
  }
}

// ---------------------------------------------------------------------------
// Encoder GRU scan: one block per direction (blockIdx.x), 32 waves.
// gi (x @ Wih^T + bih) precomputed for all steps (f32). Recurrent
// gh = h @ Whh^T via WMMA; h is bf16 in LDS + f32 in global hstate.
// Each wave owns 2 tasks; a task = (m-tile, n-triple r/z/n) so gate math is
// wave-local. Whh is pre-packed bf16.
// ---------------------------------------------------------------------------
__global__ void k_enc_scan(const float* __restrict__ giF,
                           const float* __restrict__ giB,
                           const ushort_t* __restrict__ WhhF,
                           const ushort_t* __restrict__ WhhB,
                           const float* __restrict__ bhhF,
                           const float* __restrict__ bhhB,
                           float* __restrict__ hstate,     // [2][32][512] f32
                           ushort_t* __restrict__ xsOut,   // [64,32,1024] bf16
                           int writeLast,
                           ushort_t* __restrict__ lasth) { // [32,1024] bf16
  extern __shared__ ushort_t sH[];                         // 32*512 bf16
  int dir = blockIdx.x;
  const float* gi  = dir ? giB : giF;
  const ushort_t* Whh = dir ? WhhB : WhhF;
  const float* bhh = dir ? bhhB : bhhF;
  float* hs = hstate + (size_t)dir * (32 * 512);
  int tid = threadIdx.x;
  int wave = tid >> 5, lane = tid & 31, khalf = lane >> 4, lr = lane & 15;
  for (int i = tid; i < 32 * 512; i += 1024) { sH[i] = 0; hs[i] = 0.f; }
  __syncthreads();
  for (int t = 0; t < 64; ++t) {
    int s = dir ? (63 - t) : t;
    v8f acc[2][3];
#pragma unroll
    for (int i = 0; i < 2; ++i) {
      int tau = wave * 2 + i;
      int mt = tau >> 5, trip = tau & 31;
      int mRow = mt * 16 + lr;
      v8f a0 = zero8(), a1 = zero8(), a2 = zero8();
      for (int kc = 0; kc < 512; kc += 32) {
        v16bf b = load_b_lds(sH, 512, mRow, kc, khalf);
        v16bf w0 = load_a_bf16(Whh, 512, trip * 16 + lr, kc, khalf);
        a0 = wmma32(w0, b, a0);
        v16bf w1 = load_a_bf16(Whh, 512, 512 + trip * 16 + lr, kc, khalf);
        a1 = wmma32(w1, b, a1);
        v16bf w2 = load_a_bf16(Whh, 512, 1024 + trip * 16 + lr, kc, khalf);
        a2 = wmma32(w2, b, a2);
      }
      acc[i][0] = a0; acc[i][1] = a1; acc[i][2] = a2;
    }
    __syncthreads();                       // all h reads done
#pragma unroll
    for (int i = 0; i < 2; ++i) {
      int tau = wave * 2 + i;
      int mt = tau >> 5, trip = tau & 31;
      int m = mt * 16 + lr;
      size_t rowG = (size_t)(s * 32 + m) * 1536;
#pragma unroll
      for (int j = 0; j < 8; ++j) {
        int nh = trip * 16 + khalf * 8 + j;
        float gir = gi[rowG + nh];
        float giz = gi[rowG + 512 + nh];
        float gin = gi[rowG + 1024 + nh];
        float ghr = acc[i][0][j] + bhh[nh];
        float ghz = acc[i][1][j] + bhh[512 + nh];
        float ghn = acc[i][2][j] + bhh[1024 + nh];
        float r = sigm(gir + ghr);
        float z = sigm(giz + ghz);
        float nn = tanhf(gin + r * ghn);
        float ho = hs[m * 512 + nh];
        float hv = (1.f - z) * nn + z * ho;
        ushort_t hb = f2bf(hv);
        hs[m * 512 + nh] = hv;
        sH[m * 512 + nh] = hb;
        xsOut[(size_t)(s * 32 + m) * 1024 + dir * 512 + nh] = hb;
      }
    }
    __syncthreads();                       // h updated for next step
  }
  if (writeLast) {
    for (int i = tid; i < 32 * 512; i += 1024) {
      int m = i >> 9, k = i & 511;
      lasth[m * 1024 + dir * 512 + k] = f2bf(hs[i]);
    }
  }
}

// ---------------------------------------------------------------------------
// Decoder: 2 stacked GRU cells per step, 47 steps, single 32-wave block.
// gi layer0 = Gi0emb[t] + GiA (precomputed GEMMs, GiA holds bih0).
// Layer1 input GEMM staged via gi1buf to bound register pressure.
// h1 per-step stored bf16 for the deferred classifier GEMM.
// ---------------------------------------------------------------------------
__global__ void k_dec_scan(const float* __restrict__ Gi0,
                           const float* __restrict__ GiA,
                           const ushort_t* __restrict__ Whh0,
                           const float* __restrict__ bhh0,
                           const ushort_t* __restrict__ Wih1,
                           const float* __restrict__ bih1,
                           const ushort_t* __restrict__ Whh1,
                           const float* __restrict__ bhh1,
                           const float* __restrict__ h0init,  // dec_h_init f32
                           float* __restrict__ hst0,          // [32][512]
                           float* __restrict__ hst1,          // [32][512]
                           float* __restrict__ gi1buf,        // [32][1536]
                           ushort_t* __restrict__ H1all) {    // [47,32,512]
  extern __shared__ ushort_t sD[];
  ushort_t* sH0 = sD;
  ushort_t* sH1 = sD + 32 * 512;
  int tid = threadIdx.x;
  int wave = tid >> 5, lane = tid & 31, khalf = lane >> 4, lr = lane & 15;
  for (int i = tid; i < 32 * 512; i += 1024) {
    float v = h0init[i];
    ushort_t b = f2bf(v);
    sH0[i] = b; sH1[i] = b;
    hst0[i] = v; hst1[i] = v;
  }
  __syncthreads();
  for (int t = 0; t < 47; ++t) {
    // ---- layer 0: gh0 = h0 @ Whh0^T ----
    v8f acc[2][3];
#pragma unroll
    for (int i = 0; i < 2; ++i) {
      int tau = wave * 2 + i;
      int mt = tau >> 5, trip = tau & 31;
      int mRow = mt * 16 + lr;
      v8f a0 = zero8(), a1 = zero8(), a2 = zero8();
      for (int kc = 0; kc < 512; kc += 32) {
        v16bf b = load_b_lds(sH0, 512, mRow, kc, khalf);
        v16bf w0 = load_a_bf16(Whh0, 512, trip * 16 + lr, kc, khalf);
        a0 = wmma32(w0, b, a0);
        v16bf w1 = load_a_bf16(Whh0, 512, 512 + trip * 16 + lr, kc, khalf);
        a1 = wmma32(w1, b, a1);
        v16bf w2 = load_a_bf16(Whh0, 512, 1024 + trip * 16 + lr, kc, khalf);
        a2 = wmma32(w2, b, a2);
      }
      acc[i][0] = a0; acc[i][1] = a1; acc[i][2] = a2;
    }
    __syncthreads();
#pragma unroll
    for (int i = 0; i < 2; ++i) {
      int tau = wave * 2 + i;
      int mt = tau >> 5, trip = tau & 31;
      int m = mt * 16 + lr;
      size_t rowG = (size_t)(t * 32 + m) * 1536;
      size_t rowA = (size_t)m * 1536;
#pragma unroll
      for (int j = 0; j < 8; ++j) {
        int nh = trip * 16 + khalf * 8 + j;
        float gir = Gi0[rowG + nh]        + GiA[rowA + nh];
        float giz = Gi0[rowG + 512 + nh]  + GiA[rowA + 512 + nh];
        float gin = Gi0[rowG + 1024 + nh] + GiA[rowA + 1024 + nh];
        float ghr = acc[i][0][j] + bhh0[nh];
        float ghz = acc[i][1][j] + bhh0[512 + nh];
        float ghn = acc[i][2][j] + bhh0[1024 + nh];
        float r = sigm(gir + ghr);
        float z = sigm(giz + ghz);
        float nn = tanhf(gin + r * ghn);
        float ho = hst0[m * 512 + nh];
        float hv = (1.f - z) * nn + z * ho;
        hst0[m * 512 + nh] = hv;
        sH0[m * 512 + nh] = f2bf(hv);
      }
    }
    __syncthreads();
    // ---- layer 1 pass 1: gi1 = h0_new @ Wih1^T -> gi1buf ----
#pragma unroll
    for (int i = 0; i < 2; ++i) {
      int tau = wave * 2 + i;
      int mt = tau >> 5, trip = tau & 31;
      int mRow = mt * 16 + lr;
      v8f a0 = zero8(), a1 = zero8(), a2 = zero8();
      for (int kc = 0; kc < 512; kc += 32) {
        v16bf b = load_b_lds(sH0, 512, mRow, kc, khalf);
        v16bf w0 = load_a_bf16(Wih1, 512, trip * 16 + lr, kc, khalf);
        a0 = wmma32(w0, b, a0);
        v16bf w1 = load_a_bf16(Wih1, 512, 512 + trip * 16 + lr, kc, khalf);
        a1 = wmma32(w1, b, a1);
        v16bf w2 = load_a_bf16(Wih1, 512, 1024 + trip * 16 + lr, kc, khalf);
        a2 = wmma32(w2, b, a2);
      }
#pragma unroll
      for (int j = 0; j < 8; ++j) {
        int nh = trip * 16 + khalf * 8 + j;
        gi1buf[mRow * 1536 + nh] = a0[j];
        gi1buf[mRow * 1536 + 512 + nh] = a1[j];
        gi1buf[mRow * 1536 + 1024 + nh] = a2[j];
      }
    }
    // ---- layer 1 pass 2: gh1 = h1 @ Whh1^T ----
#pragma unroll
    for (int i = 0; i < 2; ++i) {
      int tau = wave * 2 + i;
      int mt = tau >> 5, trip = tau & 31;
      int mRow = mt * 16 + lr;
      v8f a0 = zero8(), a1 = zero8(), a2 = zero8();
      for (int kc = 0; kc < 512; kc += 32) {
        v16bf b = load_b_lds(sH1, 512, mRow, kc, khalf);
        v16bf w0 = load_a_bf16(Whh1, 512, trip * 16 + lr, kc, khalf);
        a0 = wmma32(w0, b, a0);
        v16bf w1 = load_a_bf16(Whh1, 512, 512 + trip * 16 + lr, kc, khalf);
        a1 = wmma32(w1, b, a1);
        v16bf w2 = load_a_bf16(Whh1, 512, 1024 + trip * 16 + lr, kc, khalf);
        a2 = wmma32(w2, b, a2);
      }
      acc[i][0] = a0; acc[i][1] = a1; acc[i][2] = a2;
    }
    __threadfence_block();
    __syncthreads();
#pragma unroll
    for (int i = 0; i < 2; ++i) {
      int tau = wave * 2 + i;
      int mt = tau >> 5, trip = tau & 31;
      int m = mt * 16 + lr;
#pragma unroll
      for (int j = 0; j < 8; ++j) {
        int nh = trip * 16 + khalf * 8 + j;
        float gir = gi1buf[m * 1536 + nh]        + bih1[nh];
        float giz = gi1buf[m * 1536 + 512 + nh]  + bih1[512 + nh];
        float gin = gi1buf[m * 1536 + 1024 + nh] + bih1[1024 + nh];
        float ghr = acc[i][0][j] + bhh1[nh];
        float ghz = acc[i][1][j] + bhh1[512 + nh];
        float ghn = acc[i][2][j] + bhh1[1024 + nh];
        float r = sigm(gir + ghr);
        float z = sigm(giz + ghz);
        float nn = tanhf(gin + r * ghn);
        float ho = hst1[m * 512 + nh];
        float hv = (1.f - z) * nn + z * ho;
        hst1[m * 512 + nh] = hv;
        sH1[m * 512 + nh] = f2bf(hv);
        H1all[(size_t)(t * 32 + m) * 512 + nh] = f2bf(hv);
      }
    }
    __syncthreads();
  }
}

// ---------------------------------------------------------------------------
// Small helper kernels
// ---------------------------------------------------------------------------
__global__ void k_embed_enc(const int* __restrict__ x,
                            const float* __restrict__ emb,
                            ushort_t* __restrict__ xs0) {
  int idx = blockIdx.x * blockDim.x + threadIdx.x;
  if (idx >= 64 * 32 * 512) return;
  int k = idx & 511, b = (idx >> 9) & 31, s = idx >> 14;
  int tok = x[b * 64 + s];
  xs0[idx] = f2bf(emb[(size_t)tok * 512 + k]);
}

__global__ void k_embed_dec(const int* __restrict__ y,
                            const float* __restrict__ emb,
                            ushort_t* __restrict__ demb) {
  int idx = blockIdx.x * blockDim.x + threadIdx.x;
  if (idx >= 47 * 32 * 512) return;
  int k = idx & 511, b = (idx >> 9) & 31, t = idx >> 14;
  int tok = y[b * 48 + t];
  demb[idx] = f2bf(emb[(size_t)tok * 512 + k]);
}

// v_e[k] = sum_n Wa[n] * We[n][512+k]  (attention score direction)
__global__ void k_ve(const float* __restrict__ We,
                     const float* __restrict__ Wa,
                     float* __restrict__ ve) {
  int k = threadIdx.x;
  if (k >= 512) return;
  float s = 0.f;
  for (int n = 0; n < 512; ++n) s += Wa[n] * We[n * 1024 + 512 + k];
  ve[k] = s;
}

__global__ void k_scores(const ushort_t* __restrict__ enco,  // [B*S,512] bf16
                         const float* __restrict__ ve,
                         float* __restrict__ scores) {       // [B*S]
  int idx = blockIdx.x * blockDim.x + threadIdx.x;
  if (idx >= 2048) return;
  const ushort_t* e = enco + (size_t)idx * 512;
  float s = 0.f;
  for (int k = 0; k < 512; ++k) s += bf2f(e[k]) * ve[k];
  scores[idx] = s;
}

__global__ void k_attn(const float* __restrict__ scores,     // [B,S]
                       const ushort_t* __restrict__ enco,    // [B,S,512] bf16
                       ushort_t* __restrict__ attn) {        // [B,512] bf16
  int b = blockIdx.x;
  int h = threadIdx.x;
  if (h >= 512) return;
  float mx = -3.4e38f;
  for (int s = 0; s < 64; ++s) {
    float v = scores[b * 64 + s];
    mx = v > mx ? v : mx;
  }
  float den = 0.f;
  for (int s = 0; s < 64; ++s) den += expf(scores[b * 64 + s] - mx);
  float a = 0.f;
  for (int s = 0; s < 64; ++s) {
    float w = expf(scores[b * 64 + s] - mx) / den;
    a += w * bf2f(enco[((size_t)b * 64 + s) * 512 + h]);
  }
  attn[b * 512 + h] = f2bf(a);
}

__global__ void k_zero_t0(float* __restrict__ out) {         // out[:,0,:] = 0
  int idx = blockIdx.x * blockDim.x + threadIdx.x;
  if (idx >= 32 * 32000) return;
  int b = idx / 32000, n = idx - b * 32000;
  out[(size_t)b * 1536000u + n] = 0.f;
}

// ---------------------------------------------------------------------------
extern "C" void kernel_launch(void* const* d_in, const int* in_sizes, int n_in,
                              void* d_out, int out_size, void* d_ws,
                              size_t ws_size, hipStream_t stream) {
#define PF(i) ((const float*)d_in[i])
#define PI(i) ((const int*)d_in[i])
  // input order (setup_inputs dict, depth-first):
  // 0 enc_emb | 1-4 e0f Wih,Whh,bih,bhh | 5-8 e0b | 9-12 e1f | 13-16 e1b |
  // 17-20 hid W1,b1,W2,b2 | 21-24 out W1,b1,W2,b2 | 25 att_We 26 att_be
  // 27 att_Wa | 28 dec_emb | 29-32 d0 Wih,Whh,bih,bhh | 33-36 d1 |
  // 37 cls_W 38 cls_b | 39 x 40 y
  float* w = (float*)d_ws;
  size_t off = 0;                               // in floats, keep 16B aligned
  auto allocF = [&](size_t n) {
    float* p = w + off; off += (n + 3) & ~(size_t)3; return p;
  };
  auto allocU = [&](size_t n) {                 // n bf16 elements
    ushort_t* p = (ushort_t*)(w + off);
    off += ((n + 1) / 2 + 3) & ~(size_t)3; return p;
  };
  // f32 staging
  float* giF    = allocF((size_t)64 * 32 * 1536);
  float* giB    = allocF((size_t)64 * 32 * 1536);
  float* hstate = allocF((size_t)4 * 32 * 512);
  float* dechF  = allocF((size_t)32 * 512);
  float* ve     = allocF(512);
  float* scores = allocF(2048);
  float* GiA    = allocF((size_t)32 * 1536);
  float* Gi0    = allocF((size_t)47 * 32 * 1536);
  float* gi1buf = allocF((size_t)32 * 1536);
  // bf16 activations
  ushort_t* xs0B   = allocU((size_t)64 * 32 * 512);
  ushort_t* xs1B   = allocU((size_t)64 * 32 * 1024);
  ushort_t* xs2B   = allocU((size_t)64 * 32 * 1024);
  ushort_t* lasthB = allocU((size_t)32 * 1024);
  ushort_t* mlpB   = allocU((size_t)2048 * 2048);
  ushort_t* encoB  = allocU((size_t)32 * 64 * 512);
  ushort_t* attnB  = allocU((size_t)32 * 512);
  ushort_t* dembB  = allocU((size_t)47 * 32 * 512);
  ushort_t* H1allB = allocU((size_t)47 * 32 * 512);
  // bf16 weights (pre-packed once per launch)
  auto cvt = [&](const float* src, ushort_t* dst, size_t n) {
    k_cvt<<<(int)((n + 255) / 256), 256, 0, stream>>>(src, dst, (int)n);
  };
  auto wcvt = [&](int i, size_t n) {
    ushort_t* p = allocU(n); cvt(PF(i), p, n); return p;
  };
  ushort_t* e0fWih = wcvt(1,  (size_t)1536 * 512);
  ushort_t* e0fWhh = wcvt(2,  (size_t)1536 * 512);
  ushort_t* e0bWih = wcvt(5,  (size_t)1536 * 512);
  ushort_t* e0bWhh = wcvt(6,  (size_t)1536 * 512);
  ushort_t* e1fWih = wcvt(9,  (size_t)1536 * 1024);
  ushort_t* e1fWhh = wcvt(10, (size_t)1536 * 512);
  ushort_t* e1bWih = wcvt(13, (size_t)1536 * 1024);
  ushort_t* e1bWhh = wcvt(14, (size_t)1536 * 512);
  ushort_t* hidW1  = wcvt(17, (size_t)2048 * 1024);
  ushort_t* hidW2  = wcvt(19, (size_t)512 * 2048);
  ushort_t* outW1  = wcvt(21, (size_t)2048 * 1024);
  ushort_t* outW2  = wcvt(23, (size_t)512 * 2048);
  ushort_t* d0Wih  = wcvt(29, (size_t)1536 * 1024);
  ushort_t* d0Whh  = wcvt(30, (size_t)1536 * 512);
  ushort_t* d1Wih  = wcvt(33, (size_t)1536 * 512);
  ushort_t* d1Whh  = wcvt(34, (size_t)1536 * 512);
  ushort_t* clsW   = wcvt(37, (size_t)32000 * 512);
  float* out = (float*)d_out;

  auto gemm = [&](const ushort_t* W, int lda, const ushort_t* X, int ldb,
                  const float* bias, void* C, int M, int N, int K,
                  int relu, int mode, int outBf16) {
    int tasks = (M >> 5) * (N >> 4);
    int blocks = (tasks + 7) / 8;
    k_gemm<<<blocks, 256, 0, stream>>>(W, lda, X, ldb, bias, C, M, N, K,
                                       relu, mode, outBf16);
  };

  // ---- encoder ----
  k_embed_enc<<<(64 * 32 * 512 + 255) / 256, 256, 0, stream>>>(PI(39), PF(0),
                                                               xs0B);
  gemm(e0fWih, 512, xs0B, 512, PF(3), giF, 2048, 1536, 512, 0, 0, 0);
  gemm(e0bWih, 512, xs0B, 512, PF(7), giB, 2048, 1536, 512, 0, 0, 0);
  k_enc_scan<<<2, 1024, 32 * 512 * 2, stream>>>(giF, giB, e0fWhh, e0bWhh,
                                                PF(4), PF(8), hstate, xs1B, 0,
                                                (ushort_t*)nullptr);
  gemm(e1fWih, 1024, xs1B, 1024, PF(11), giF, 2048, 1536, 1024, 0, 0, 0);
  gemm(e1bWih, 1024, xs1B, 1024, PF(15), giB, 2048, 1536, 1024, 0, 0, 0);
  k_enc_scan<<<2, 1024, 32 * 512 * 2, stream>>>(giF, giB, e1fWhh, e1bWhh,
                                                PF(12), PF(16), hstate, xs2B,
                                                1, lasthB);
  // ---- bridge MLPs ----
  gemm(hidW1, 1024, lasthB, 1024, PF(18), mlpB, 32, 2048, 1024, 1, 0, 1);
  gemm(hidW2, 2048, mlpB, 2048, PF(20), dechF, 32, 512, 2048, 0, 0, 0);
  gemm(outW1, 1024, xs2B, 1024, PF(22), mlpB, 2048, 2048, 1024, 1, 0, 1);
  gemm(outW2, 2048, mlpB, 2048, PF(24), encoB, 2048, 512, 2048, 0, 1, 1);
  // ---- attention (decoder-state independent: linear energy => q cancels) ----
  k_ve<<<1, 512, 0, stream>>>(PF(25), PF(27), ve);
  k_scores<<<8, 256, 0, stream>>>(encoB, ve, scores);
  k_attn<<<32, 512, 0, stream>>>(scores, encoB, attnB);
  // ---- decoder input precompute ----
  k_embed_dec<<<(47 * 32 * 512 + 255) / 256, 256, 0, stream>>>(PI(40), PF(28),
                                                               dembB);
  gemm(d0Wih, 1024, dembB, 512, (const float*)nullptr, Gi0, 1504, 1536, 512,
       0, 0, 0);                                 // emb part of d0.Wih
  gemm(d0Wih + 512, 1024, attnB, 512, PF(31), GiA, 32, 1536, 512, 0, 0, 0);
  // ---- decoder recurrence ----
  k_dec_scan<<<1, 1024, 2 * 32 * 512 * 2, stream>>>(
      Gi0, GiA, d0Whh, PF(32), d1Wih, PF(35), d1Whh, PF(36), dechF,
      hstate + 2 * 32 * 512, hstate + 3 * 32 * 512, gi1buf, H1allB);
  // ---- classifier (deferred, one big GEMM) + zero t=0 slice ----
  k_zero_t0<<<(32 * 32000 + 255) / 256, 256, 0, stream>>>(out);
  gemm(clsW, 512, H1allB, 512, PF(38), out, 1504, 32000, 512, 0, 2, 0);
#undef PF
#undef PI
}
